// BSMamba3Block_69475390980342
// MI455X (gfx1250) — matrix operand order repaired and checked by reading, not compiled, not run on hardware
//
#include <hip/hip_runtime.h>
#include <cstddef>

// ---------------------------------------------------------------------------
// Types for CDNA5 WMMA / TDM / async loads
// ---------------------------------------------------------------------------
typedef __attribute__((ext_vector_type(16))) __bf16         v16bf;
typedef __attribute__((ext_vector_type(16))) unsigned short v16u;
typedef __attribute__((ext_vector_type(8)))  float          v8f;
typedef __attribute__((ext_vector_type(4)))  unsigned int   v4u;
typedef __attribute__((ext_vector_type(8)))  int            v8i;
typedef __attribute__((ext_vector_type(4)))  int            v4i;

// Model constants
#define BB   2
#define TT   256
#define KK   8
#define DD   256
#define HH   4
#define PP   64
#define NN   128
#define MM   4096   // = BB*TT*KK = BB*KK*TT

// Feature detection (compile-safe: degrade gracefully)
#if __has_builtin(__builtin_amdgcn_global_load_async_to_lds_b128) && \
    __has_builtin(__builtin_amdgcn_s_wait_asynccnt)
#define GEMM_ASYNC 1
#else
#define GEMM_ASYNC 0
#endif
#if __has_builtin(__builtin_amdgcn_tensor_load_to_lds) && \
    __has_builtin(__builtin_amdgcn_s_wait_tensorcnt)
#define GEMM_TDM 1
#else
#define GEMM_TDM 0
#endif

#if GEMM_ASYNC
// async-load builtin takes typed, address-space-qualified pointers:
//   (v4i addrspace(1)* gsrc, v4i addrspace(3)* ldst, imm offset, imm cpol)
typedef __attribute__((address_space(1))) v4i* gv4i_p;
typedef __attribute__((address_space(3))) v4i* lv4i_p;
#endif

// ---------------------------------------------------------------------------
// Helpers
// ---------------------------------------------------------------------------
__device__ __forceinline__ unsigned short f2bf(float f) {
  unsigned u = __builtin_bit_cast(unsigned, f);
  u += 0x7FFFu + ((u >> 16) & 1u);      // RNE
  return (unsigned short)(u >> 16);
}
__device__ __forceinline__ float bf2f(unsigned short s) {
  unsigned u = ((unsigned)s) << 16;
  return __builtin_bit_cast(float, u);
}
__device__ __forceinline__ float siluf(float x) { return x / (1.0f + __expf(-x)); }

#if GEMM_TDM
// 2D TDM load: tensor [rows, rowlen] (data_size=2B), tile [tile_rows, tile_k],
// LDS destination padded +16B per 64B row so LDS pitch = 80B (LDT=40 ushorts).
__device__ __forceinline__ void tdm_load_tile_2d(const void* gsrc, unsigned lds_off,
                                                 unsigned rows, unsigned rowlen,
                                                 unsigned row_stride,
                                                 unsigned tile_k, unsigned tile_rows) {
  unsigned long long ga = (unsigned long long)(size_t)gsrc;
  v4u g0;
  g0[0] = 1u;                                            // count=1, user mode
  g0[1] = lds_off;                                       // lds_addr (bytes)
  g0[2] = (unsigned)(ga & 0xFFFFFFFFu);                  // global_addr[31:0]
  g0[3] = (unsigned)((ga >> 32) & 0x01FFFFFFu) | (2u << 30);  // addr[56:32] | type=2
  // group1: data_size=2B (code 1), pad_enable, pad_interval=16 DWORDs (code 3),
  // pad_amount=4 DWORDs (code 3)  -> pitch 64B + 16B = 80B
  unsigned w0 = (1u << 16) | (1u << 20) | (3u << 22) | (3u << 25);
  v8i g1;
  g1[0] = (int)w0;
  g1[1] = (int)((rowlen & 0xFFFFu) << 16);               // tensor_dim0[15:0]
  g1[2] = (int)(((rowlen >> 16) & 0xFFFFu) | ((rows & 0xFFFFu) << 16));
  g1[3] = (int)(((rows >> 16) & 0xFFFFu) | ((tile_k & 0xFFFFu) << 16)); // tile_dim0
  g1[4] = (int)(tile_rows & 0xFFFFu);                    // tile_dim1 (tile_dim2=0)
  g1[5] = (int)row_stride;                               // tensor_dim0_stride[31:0]
  g1[6] = 0;                                             // stride0[47:32]|stride1[15:0]
  g1[7] = 0;
  v4i z4 = {0, 0, 0, 0};
#if defined(__clang_major__) && __clang_major__ >= 23
  v8i z8 = {0, 0, 0, 0, 0, 0, 0, 0};
  __builtin_amdgcn_tensor_load_to_lds(g0, g1, z4, z4, z8, 0);
#else
  __builtin_amdgcn_tensor_load_to_lds(g0, g1, z4, z4, 0);
#endif
}
#endif

// ---------------------------------------------------------------------------
// Weight convert: src f32 [R,C]; to_nk==1 -> dst bf16 [C,R] (i.e. [N,K] layout
// from a [K,N] weight); to_nk==0 -> dst bf16 same layout (already [N,K]).
// ---------------------------------------------------------------------------
__global__ void k_wconv(const float* __restrict__ src, unsigned short* __restrict__ dst,
                        int R, int C, int to_nk) {
  int idx = blockIdx.x * 256 + threadIdx.x;
  if (idx >= R * C) return;
  int r = idx / C, c = idx % C;
  unsigned short v = f2bf(src[idx]);
  if (to_nk) dst[(size_t)c * R + r] = v;
  else       dst[idx] = v;
}

__global__ void k_cvt_bf16(const float* __restrict__ src, unsigned short* __restrict__ dst, int n) {
  int idx = blockIdx.x * 256 + threadIdx.x;
  if (idx < n) dst[idx] = f2bf(src[idx]);
}

// ---------------------------------------------------------------------------
// LayerNorm over D=256 -> bf16.  permuteA: read z[B,T,K,D] as u[bk*T+t, d]
// ---------------------------------------------------------------------------
__global__ __launch_bounds__(256) void k_ln_bf16(const float* __restrict__ in,
                                                 const float* __restrict__ g,
                                                 const float* __restrict__ b,
                                                 unsigned short* __restrict__ out,
                                                 int permuteA) {
  __shared__ float red[256];
  const int r = blockIdx.x;
  const int d = threadIdx.x;
  size_t in_off;
  if (permuteA) {
    int t  = r & (TT - 1);
    int bk = r >> 8;
    int k  = bk & (KK - 1);
    int bb = bk >> 3;
    in_off = ((size_t)((bb * TT + t) * KK + k)) * DD + d;
  } else {
    in_off = (size_t)r * DD + d;
  }
  float x = in[in_off];
  red[d] = x; __syncthreads();
  for (int s = 128; s > 0; s >>= 1) { if (d < s) red[d] += red[d + s]; __syncthreads(); }
  float mean = red[0] * (1.0f / DD);
  __syncthreads();
  float c = x - mean;
  red[d] = c * c; __syncthreads();
  for (int s = 128; s > 0; s >>= 1) { if (d < s) red[d] += red[d + s]; __syncthreads(); }
  float var = red[0] * (1.0f / DD);
  float y = c * rsqrtf(var + 1e-5f) * g[d] + b[d];
  out[(size_t)r * DD + d] = f2bf(y);
}

// ---------------------------------------------------------------------------
// WMMA GEMM:  C[M,N] (f32) = A[M,K](bf16,row) * Wt[N,K](bf16,row)  (+bias)(+=)
// Block 256 thr = 8 waves; block tile 64x64; wave tile 16x32; K-step 32.
// A-tile staged by per-thread async b128 loads; B-tile staged by one TDM
// descriptor (pad -> 80B pitch); sync fallback if builtins unavailable.
// ---------------------------------------------------------------------------
#define LDT 40   // ushort pitch of both LDS tiles (80 bytes)
__global__ __launch_bounds__(256) void k_gemm_bf16(const unsigned short* __restrict__ A,
                                                   const unsigned short* __restrict__ W,
                                                   float* __restrict__ C,
                                                   const float* __restrict__ bias,
                                                   int M, int N, int Kd, int accum) {
  __shared__ unsigned short As[64 * LDT];   // 64 m-rows x 32 k (+pad)
  __shared__ unsigned short Bs[64 * LDT];   // 64 n-rows x 32 k (+pad)
  const int tid   = threadIdx.x;
  const int wave  = tid >> 5;
  const int lane  = tid & 31;
  const int row16 = lane & 15;
  const int hi    = lane >> 4;
  const int wm    = wave & 3;     // M sub-tile 0..3
  const int wn    = wave >> 2;    // N half 0..1
  const int m0 = blockIdx.y * 64;
  const int n0 = blockIdx.x * 64;

  v8f acc0 = {}; v8f acc1 = {};

  const int tr = tid >> 2;          // 0..63 tile row
  const int tc = (tid & 3) * 8;     // k sub-chunk (8 ushorts = 16B)
  const unsigned short* gA = A + (size_t)(m0 + tr) * Kd + tc;
  const unsigned short* gB = W + (size_t)(n0 + tr) * Kd + tc;
  unsigned short* lA = As + tr * LDT + tc;
  unsigned short* lB = Bs + tr * LDT + tc;
#if GEMM_TDM
  const unsigned bs_off = (unsigned)(size_t)Bs;   // LDS byte offset of Bs
#endif

  for (int k0 = 0; k0 < Kd; k0 += 32) {
    // ---- stage A tile ----
#if GEMM_ASYNC
    __builtin_amdgcn_global_load_async_to_lds_b128((gv4i_p)(void*)(gA + k0),
                                                   (lv4i_p)(void*)lA, 0, 0);
#else
    *(uint4*)lA = *(const uint4*)(gA + k0);
#endif
    // ---- stage B tile ----
#if GEMM_TDM
    if (tid < 32)
      tdm_load_tile_2d(W + (size_t)n0 * Kd + k0, bs_off,
                       (unsigned)N, (unsigned)Kd, (unsigned)Kd, 32u, 64u);
#elif GEMM_ASYNC
    __builtin_amdgcn_global_load_async_to_lds_b128((gv4i_p)(void*)(gB + k0),
                                                   (lv4i_p)(void*)lB, 0, 0);
#else
    *(uint4*)lB = *(const uint4*)(gB + k0);
#endif
    if (k0 + 32 < Kd) {
      __builtin_prefetch(gA + k0 + 32, 0, 1);
      __builtin_prefetch(gB + k0 + 32, 0, 1);
    }
#if GEMM_ASYNC
    __builtin_amdgcn_s_wait_asynccnt(0);
#endif
#if GEMM_TDM
    __builtin_amdgcn_s_wait_tensorcnt(0);
#endif
    __syncthreads();

    // ---- A fragment: two b128 LDS loads (K runs hi*8+{0..7} and +16..23) ----
    const int arow = (wm * 16 + row16) * LDT + hi * 8;
    union { v16u v; uint4 q[2]; } ua;
    ua.q[0] = *(const uint4*)(As + arow);
    ua.q[1] = *(const uint4*)(As + arow + 16);
    v16bf afrag = __builtin_bit_cast(v16bf, ua.v);

    // ---- B fragments: K runs hi*8+{0..7} and +16..23 per column ----
    const int b0 = (wn * 32 + row16) * LDT + hi * 8;
    const int b1 = b0 + 16 * LDT;
    union { v16u v; uint4 q[2]; } ub0, ub1;
    ub0.q[0] = *(const uint4*)(Bs + b0);
    ub0.q[1] = *(const uint4*)(Bs + b0 + 16);
    ub1.q[0] = *(const uint4*)(Bs + b1);
    ub1.q[1] = *(const uint4*)(Bs + b1 + 16);
    v16bf bf0 = __builtin_bit_cast(v16bf, ub0.v);
    v16bf bf1 = __builtin_bit_cast(v16bf, ub1.v);

    acc0 = __builtin_amdgcn_wmma_f32_16x16x32_bf16(false, afrag, false, bf0,
                                                   (short)0, acc0, false, false);
    acc1 = __builtin_amdgcn_wmma_f32_16x16x32_bf16(false, afrag, false, bf1,
                                                   (short)0, acc1, false, false);
    __syncthreads();
  }

  // C/D layout: element i -> row M = i + 8*hi, col N = lane%16
#pragma unroll
  for (int i = 0; i < 8; ++i) {
    int m  = m0 + wm * 16 + hi * 8 + i;
    int nA = n0 + wn * 32 + row16;
    int nB = nA + 16;
    float vA = acc0[i], vB = acc1[i];
    if (bias) { vA += bias[nA]; vB += bias[nB]; }
    size_t iA = (size_t)m * N + nA, iB = (size_t)m * N + nB;
    if (accum) { C[iA] += vA; C[iB] += vB; }
    else       { C[iA]  = vA; C[iB]  = vB; }
  }
}

// ---------------------------------------------------------------------------
// Causal depthwise conv (kernel 4) + SiLU.  x,xh: [4096,256] rows = bk*T+t
// ---------------------------------------------------------------------------
__global__ __launch_bounds__(256) void k_conv_silu(const float* __restrict__ x,
                                                   const float* __restrict__ convw,
                                                   float* __restrict__ xh) {
  const int m = blockIdx.x;
  const int d = threadIdx.x;
  const int t = m & (TT - 1);
  float acc = 0.f;
#pragma unroll
  for (int kk = 0; kk < 4; ++kk) {
    int tt = t - 3 + kk;
    if (tt >= 0) acc += x[(size_t)(m - 3 + kk) * DD + d] * convw[d * 4 + kk];
  }
  xh[(size_t)m * DD + d] = siluf(acc);
}

// ---------------------------------------------------------------------------
// dt / decay precompute
// ---------------------------------------------------------------------------
__global__ __launch_bounds__(256) void k_dt(const unsigned short* __restrict__ u_bf,
                                            const float* __restrict__ Wdt,
                                            const float* __restrict__ dt_bias,
                                            const float* __restrict__ A_log,
                                            const float* __restrict__ theta,
                                            float* __restrict__ dtv,
                                            float* __restrict__ arv,
                                            float* __restrict__ aiv) {
  int idx = blockIdx.x * 256 + threadIdx.x;   // 4096*4 total
  int m = idx >> 2, h = idx & 3;
  float acc = 0.f;
  for (int d = 0; d < DD; ++d)
    acc += bf2f(u_bf[(size_t)m * DD + d]) * Wdt[d * HH + h];
  float s  = acc + dt_bias[h];
  float dt = (s > 20.f) ? s : log1pf(__expf(s));
  float A  = -__expf(A_log[h]);
  float mg = __expf(dt * A);
  float ph = dt * theta[h];
  dtv[idx] = dt;
  arv[idx] = mg * __cosf(ph);
  aiv[idx] = mg * __sinf(ph);
}

// ---------------------------------------------------------------------------
// Cross-head mix:  Bmix[m, h*128+n] = sum_g (U V^T)[h,g] * Bm[m, g*128+n]
// ---------------------------------------------------------------------------
__global__ __launch_bounds__(256) void k_bmix(const float* __restrict__ Bm,
                                              const float* __restrict__ U,
                                              const float* __restrict__ V,
                                              float* __restrict__ Bmix) {
  int idx = blockIdx.x * 256 + threadIdx.x;   // 4096*512 total
  int m = idx >> 9, c = idx & 511;
  int h = c >> 7, n = c & 127;
  float acc = 0.f;
#pragma unroll
  for (int g = 0; g < HH; ++g) {
    float mhg = 0.f;
#pragma unroll
    for (int r = 0; r < HH; ++r) mhg += U[h * HH + r] * V[g * HH + r];
    acc += mhg * Bm[(size_t)m * 512 + g * NN + n];
  }
  Bmix[idx] = acc;
}

// ---------------------------------------------------------------------------
// Complex MIMO selective scan. One block per (head, band); state in registers.
// ---------------------------------------------------------------------------
__global__ __launch_bounds__(256) void k_scan(const float* __restrict__ xh,
                                              const float* __restrict__ Bm,
                                              const float* __restrict__ Cm,
                                              const float* __restrict__ dtv,
                                              const float* __restrict__ arv,
                                              const float* __restrict__ aiv,
                                              float* __restrict__ ys) {
  const int h  = blockIdx.x;   // 0..3
  const int bk = blockIdx.y;   // 0..15
  const int tid = threadIdx.x;
  const int p   = tid & 63;
  const int nb  = tid >> 6;    // 0..3 (n blocks of 32)
  __shared__ float bt_s[NN], ct_s[NN], xt_s[PP], ypart[4][PP];
  float hre[32], him[32];
#pragma unroll
  for (int i = 0; i < 32; ++i) { hre[i] = 0.f; him[i] = 0.f; }

  for (int t = 0; t < TT; ++t) {
    const size_t m = (size_t)bk * TT + t;
    if (tid < NN) {
      bt_s[tid] = Bm[m * 512 + h * NN + tid];
      ct_s[tid] = Cm[m * 512 + h * NN + tid];
    } else if (tid < NN + PP) {
      xt_s[tid - NN] = xh[m * DD + h * PP + (tid - NN)];
    }
    __syncthreads();
    const float dt = dtv[m * HH + h];
    const float ar = arv[m * HH + h];
    const float ai = aiv[m * HH + h];
    const float xt = xt_s[p];
    float acc = 0.f;
#pragma unroll
    for (int i = 0; i < 32; ++i) {
      const int n = nb * 32 + i;
      const float dBx = dt * bt_s[n] * xt;
      const float nr = ar * hre[i] - ai * him[i] + dBx;
      const float ni = ar * him[i] + ai * hre[i];
      hre[i] = nr; him[i] = ni;
      acc += ct_s[n] * nr;
    }
    ypart[nb][p] = acc;
    __syncthreads();
    if (tid < PP)
      ys[m * DD + h * PP + tid] =
          ypart[0][tid] + ypart[1][tid] + ypart[2][tid] + ypart[3][tid];
    __syncthreads();
  }
}

// ---------------------------------------------------------------------------
// Gate: y2 = bf16((ys + D_skip[h]*xh) * silu(zgate))
// ---------------------------------------------------------------------------
__global__ __launch_bounds__(256) void k_ygate(const float* __restrict__ ys,
                                               const float* __restrict__ xh,
                                               const float* __restrict__ Dskip,
                                               const float* __restrict__ zgate,
                                               unsigned short* __restrict__ y2) {
  const size_t i = (size_t)blockIdx.x * DD + threadIdx.x;
  const int h = threadIdx.x >> 6;
  float v = (ys[i] + Dskip[h] * xh[i]) * siluf(zgate[i]);
  y2[i] = f2bf(v);
}

// d_out[b,t,k,d] += src[(bk*T+t), d]
__global__ __launch_bounds__(256) void k_remap_add(const float* __restrict__ src,
                                                   float* __restrict__ out) {
  const int m = blockIdx.x;
  const int d = threadIdx.x;
  const int t = m & (TT - 1);
  const int bk = m >> 8;
  const int k = bk & (KK - 1);
  const int bb = bk >> 3;
  out[((size_t)((bb * TT + t) * KK + k)) * DD + d] += src[(size_t)m * DD + d];
}

// ---------------------------------------------------------------------------
// Windowed attention: per (b*t, window, head): 4x4 softmax over WIN bands
// ---------------------------------------------------------------------------
__global__ __launch_bounds__(64) void k_attn(const float* __restrict__ qkv,
                                             float* __restrict__ o) {
  const int bid = blockIdx.x;       // 512 * 2 * 4
  const int h  = bid & 3;
  const int w  = (bid >> 2) & 1;
  const int m2 = bid >> 3;          // 0..511
  const int d  = threadIdx.x;       // 0..63
  __shared__ float qs[4][PP], ks[4][PP], vs[4][PP], sc[4][4], aw[4][4];
#pragma unroll
  for (int i = 0; i < 4; ++i) {
    size_t row = (size_t)m2 * KK + w * 4 + i;
    qs[i][d] = qkv[row * 768 + h * PP + d];
    ks[i][d] = qkv[row * 768 + 256 + h * PP + d];
    vs[i][d] = qkv[row * 768 + 512 + h * PP + d];
  }
  __syncthreads();
  if (d < 16) {
    int qi = d >> 2, ki = d & 3;
    float s = 0.f;
    for (int e = 0; e < PP; ++e) s += qs[qi][e] * ks[ki][e];
    sc[qi][ki] = s * 0.125f;          // 1/sqrt(64)
  }
  __syncthreads();
  if (d < 4) {
    float mx = sc[d][0];
#pragma unroll
    for (int ki = 1; ki < 4; ++ki) mx = fmaxf(mx, sc[d][ki]);
    float e[4], sum = 0.f;
#pragma unroll
    for (int ki = 0; ki < 4; ++ki) { e[ki] = __expf(sc[d][ki] - mx); sum += e[ki]; }
#pragma unroll
    for (int ki = 0; ki < 4; ++ki) aw[d][ki] = e[ki] / sum;
  }
  __syncthreads();
#pragma unroll
  for (int qi = 0; qi < 4; ++qi) {
    float acc = 0.f;
#pragma unroll
    for (int ki = 0; ki < 4; ++ki) acc += aw[qi][ki] * vs[ki][d];
    size_t row = (size_t)m2 * KK + w * 4 + qi;
    o[row * DD + h * PP + d] = acc;
  }
}

// FFN gate: gu = bf16(silu(g) * u)
__global__ void k_silu_mul(const float* __restrict__ g, const float* __restrict__ u,
                           unsigned short* __restrict__ gu, int n) {
  int idx = blockIdx.x * 256 + threadIdx.x;
  if (idx < n) gu[idx] = f2bf(siluf(g[idx]) * u[idx]);
}

// ---------------------------------------------------------------------------
// Host orchestration
// ---------------------------------------------------------------------------
extern "C" void kernel_launch(void* const* d_in, const int* in_sizes, int n_in,
                              void* d_out, int out_size, void* d_ws, size_t ws_size,
                              hipStream_t stream) {
  const float* z        = (const float*)d_in[0];
  const float* ln1_g    = (const float*)d_in[1];
  const float* ln1_b    = (const float*)d_in[2];
  const float* Wx       = (const float*)d_in[3];
  const float* conv_w   = (const float*)d_in[4];
  const float* Wz       = (const float*)d_in[5];
  const float* Wb       = (const float*)d_in[6];
  const float* Wc       = (const float*)d_in[7];
  const float* Wdt      = (const float*)d_in[8];
  const float* dt_bias  = (const float*)d_in[9];
  const float* A_log    = (const float*)d_in[10];
  const float* theta    = (const float*)d_in[11];
  const float* D_skip   = (const float*)d_in[12];
  const float* mimo_U   = (const float*)d_in[13];
  const float* mimo_V   = (const float*)d_in[14];
  const float* Wout     = (const float*)d_in[15];
  const float* ln2_g    = (const float*)d_in[16];
  const float* ln2_b    = (const float*)d_in[17];
  const float* attn_in_w  = (const float*)d_in[18];
  const float* attn_in_b  = (const float*)d_in[19];
  const float* attn_out_w = (const float*)d_in[20];
  const float* attn_out_b = (const float*)d_in[21];
  const float* ln3_g    = (const float*)d_in[22];
  const float* ln3_b    = (const float*)d_in[23];
  const float* Wg       = (const float*)d_in[24];
  const float* Wu       = (const float*)d_in[25];
  const float* Wd       = (const float*)d_in[26];

  char* ws = (char*)d_ws;
  size_t off = 0;
  auto alloc = [&](size_t bytes) -> char* {
    char* p = ws + off;
    off += (bytes + 255) & ~(size_t)255;
    return p;
  };

  // Persistent bf16 weights, all stored as [N, K] (K contiguous)
  unsigned short* wWx  = (unsigned short*)alloc((size_t)256 * 256 * 2);
  unsigned short* wWb  = (unsigned short*)alloc((size_t)256 * 512 * 2);
  unsigned short* wWc  = (unsigned short*)alloc((size_t)256 * 512 * 2);
  unsigned short* wWz  = (unsigned short*)alloc((size_t)256 * 256 * 2);
  unsigned short* wWo  = (unsigned short*)alloc((size_t)256 * 256 * 2);
  unsigned short* wQKV = (unsigned short*)alloc((size_t)256 * 768 * 2);
  unsigned short* wAO  = (unsigned short*)alloc((size_t)256 * 256 * 2);
  unsigned short* wWg  = (unsigned short*)alloc((size_t)256 * 1024 * 2);
  unsigned short* wWu  = (unsigned short*)alloc((size_t)256 * 1024 * 2);
  unsigned short* wWd  = (unsigned short*)alloc((size_t)1024 * 256 * 2);

  // Reusable large regions
  char* R1 = alloc((size_t)16 << 20);  // Bm raw | qkv | g
  char* R2 = alloc((size_t)16 << 20);  // Cm     | o   | u (ffn)
  char* R3 = alloc((size_t)8 << 20);   // Bmix   | zn_bf+o_bf | gu_bf
  char* R4 = alloc((size_t)4 << 20);   // x      | zf_bf
  float* xh    = (float*)alloc((size_t)MM * DD * 4);
  float* zgate = (float*)alloc((size_t)MM * DD * 4);
  float* ys    = (float*)alloc((size_t)MM * DD * 4);   // also reused as woutres
  unsigned short* u_bf  = (unsigned short*)alloc((size_t)MM * DD * 2);
  unsigned short* y2_bf = (unsigned short*)alloc((size_t)MM * DD * 2);
  float* dtv = (float*)alloc((size_t)MM * HH * 4);
  float* arv = (float*)alloc((size_t)MM * HH * 4);
  float* aiv = (float*)alloc((size_t)MM * HH * 4);

  // d_out starts as the residual stream z
  (void)hipMemcpyAsync(d_out, z, (size_t)out_size * sizeof(float),
                       hipMemcpyDeviceToDevice, stream);

  // --- weight conversion (to [N,K] bf16) ---
  auto wc = [&](const float* s, unsigned short* d, int R, int C, int to_nk) {
    int n = R * C;
    k_wconv<<<(n + 255) / 256, 256, 0, stream>>>(s, d, R, C, to_nk);
  };
  wc(Wx, wWx, 256, 256, 1);          // [K,N] -> [N,K]
  wc(Wb, wWb, 256, 512, 1);
  wc(Wc, wWc, 256, 512, 1);
  wc(Wz, wWz, 256, 256, 1);
  wc(Wout, wWo, 256, 256, 1);
  wc(attn_in_w, wQKV, 768, 256, 0);  // already [N,K]
  wc(attn_out_w, wAO, 256, 256, 0);  // already [N,K]
  wc(Wg, wWg, 256, 1024, 1);
  wc(Wu, wWu, 256, 1024, 1);
  wc(Wd, wWd, 1024, 256, 1);

  auto gemm = [&](const unsigned short* A, const unsigned short* W, float* C,
                  const float* bias, int M, int N, int Kd, int accum) {
    dim3 grid(N / 64, M / 64);
    k_gemm_bf16<<<grid, 256, 0, stream>>>(A, W, C, bias, M, N, Kd, accum);
  };

  // ================= Stage A: Mamba =================
  k_ln_bf16<<<MM, 256, 0, stream>>>((const float*)d_out, ln1_g, ln1_b, u_bf, 1);

  float* xbuf  = (float*)R4;
  float* BmRaw = (float*)R1;
  float* Cm    = (float*)R2;
  float* Bmix  = (float*)R3;

  gemm(u_bf, wWx, xbuf, nullptr, MM, 256, 256, 0);
  gemm(u_bf, wWb, BmRaw, nullptr, MM, 512, 256, 0);
  gemm(u_bf, wWc, Cm, nullptr, MM, 512, 256, 0);
  gemm(u_bf, wWz, zgate, nullptr, MM, 256, 256, 0);

  k_dt<<<(MM * HH) / 256, 256, 0, stream>>>(u_bf, Wdt, dt_bias, A_log, theta,
                                            dtv, arv, aiv);
  k_conv_silu<<<MM, 256, 0, stream>>>(xbuf, conv_w, xh);
  k_bmix<<<(MM * 512) / 256, 256, 0, stream>>>(BmRaw, mimo_U, mimo_V, Bmix);

  k_scan<<<dim3(HH, BB * KK), 256, 0, stream>>>(xh, Bmix, Cm, dtv, arv, aiv, ys);

  k_ygate<<<MM, 256, 0, stream>>>(ys, xh, D_skip, zgate, y2_bf);
  float* woutres = ys;                       // ys consumed; reuse
  gemm(y2_bf, wWo, woutres, nullptr, MM, 256, 256, 0);
  k_remap_add<<<MM, 256, 0, stream>>>(woutres, (float*)d_out);

  // ================= Stage B: attention =================
  unsigned short* zn_bf = (unsigned short*)R3;
  unsigned short* o_bf  = (unsigned short*)(R3 + ((size_t)4 << 20));
  float* qkv  = (float*)R1;
  float* obuf = (float*)R2;

  k_ln_bf16<<<MM, 256, 0, stream>>>((const float*)d_out, ln2_g, ln2_b, zn_bf, 0);
  gemm(zn_bf, wQKV, qkv, attn_in_b, MM, 768, 256, 0);
  k_attn<<<BB * TT * 2 * HH, 64, 0, stream>>>(qkv, obuf);
  k_cvt_bf16<<<(MM * DD) / 256, 256, 0, stream>>>(obuf, o_bf, MM * DD);
  gemm(o_bf, wAO, (float*)d_out, attn_out_b, MM, 256, 256, 1);  // residual add

  // ================= Stage C: SwiGLU FFN =================
  unsigned short* zf_bf = (unsigned short*)R4;
  float* gbuf = (float*)R1;
  float* ubuf = (float*)R2;
  unsigned short* gu_bf = (unsigned short*)R3;

  k_ln_bf16<<<MM, 256, 0, stream>>>((const float*)d_out, ln3_g, ln3_b, zf_bf, 0);
  gemm(zf_bf, wWg, gbuf, nullptr, MM, 1024, 256, 0);
  gemm(zf_bf, wWu, ubuf, nullptr, MM, 1024, 256, 0);
  k_silu_mul<<<(MM * 1024) / 256, 256, 0, stream>>>(gbuf, ubuf, gu_bf, MM * 1024);
  gemm(gu_bf, wWd, (float*)d_out, nullptr, MM, 256, 1024, 1);   // residual add

  (void)in_sizes; (void)n_in; (void)ws_size;
}
